// GlobalGraphSAGE_70660801953850
// MI455X (gfx1250) — matrix operand order repaired and checked by reading, not compiled, not run on hardware
//
#include <hip/hip_runtime.h>
#include <hip/hip_bf16.h>
#include <math.h>

typedef __attribute__((ext_vector_type(16))) _Float16 v16h;
typedef __attribute__((ext_vector_type(4)))  _Float16 h4;
typedef __attribute__((ext_vector_type(8)))  float    v8f;
typedef __attribute__((ext_vector_type(4)))  int      i4;

#define NODES 2048
#define EDGES 131072
#define INDIM 512
#define HIDD  2048
#define NCLS  16

// gfx1250 async global->LDS copy (guarded; fall back to b128 register copy)
#if defined(__has_builtin)
#if __has_builtin(__builtin_amdgcn_global_load_async_to_lds_b128)
#define HAVE_ASYNC_LDS 1
#endif
#endif
#ifndef HAVE_ASYNC_LDS
#define HAVE_ASYNC_LDS 0
#endif

// ---------------------------------------------------------------------------
// Aggregation kernels (memory-bound; atomics land in 192MB L2)
// ---------------------------------------------------------------------------
__global__ void k_degree(const int* __restrict__ dst, float* __restrict__ deg, int E) {
    int e = blockIdx.x * blockDim.x + threadIdx.x;
    if (e < E) atomicAdd(&deg[dst[e]], 1.0f);
}

__global__ void k_scatter_add(const float* __restrict__ x,
                              const int* __restrict__ src,
                              const int* __restrict__ dst,
                              float* __restrict__ sum,
                              int E, int F) {
    int i = blockIdx.x * blockDim.x + threadIdx.x;
    int chunks = F >> 2;
    int total = E * chunks;
    if (i >= total) return;
    int e = i / chunks;
    int f = (i - e * chunks) << 2;
    int s = src[e];
    int d = dst[e];
    const float4 v = *reinterpret_cast<const float4*>(&x[(size_t)s * F + f]);
    float* out = &sum[(size_t)d * F + f];
    atomicAdd(out + 0, v.x);
    atomicAdd(out + 1, v.y);
    atomicAdd(out + 2, v.z);
    atomicAdd(out + 3, v.w);
}

__global__ void k_scale(float* __restrict__ sum, const float* __restrict__ deg,
                        int Nn, int F) {
    int i = blockIdx.x * blockDim.x + threadIdx.x;
    if (i >= Nn * F) return;
    int row = i / F;
    sum[i] *= 1.0f / fmaxf(deg[row], 1.0f);
}

// ---------------------------------------------------------------------------
// Fragment packing (f32 row-major -> f16 WMMA fragment order).
// A operand: per (m-tile of 16, k-tile of 32) block of 512 halfs:
//   offset = (tm*(K/32)+tk)*512 + (m%16 + 16*grp)*16 + h
//   where for k in [0,32): hi=k>=16, k4=k&15, grp=k4>>3, v=(k4&7)>>1 + hi*4,
//   h = 2v + (k&1).  4 consecutive k (4-aligned) -> 4 contiguous halfs.
// ---------------------------------------------------------------------------
__global__ void k_pack_a(const float* __restrict__ X, _Float16* __restrict__ P,
                         int Mrows, int K) {
    int i = blockIdx.x * blockDim.x + threadIdx.x;
    int kg = K >> 2;
    int total = Mrows * kg;
    if (i >= total) return;
    int m = i / kg;
    int k0 = (i - m * kg) << 2;
    const float4 v = *reinterpret_cast<const float4*>(&X[(size_t)m * K + k0]);
    int tm = m >> 4, l16 = m & 15;
    int tk = k0 >> 5, kk = k0 & 31;
    int hi = kk >> 4, k4 = kk & 15;
    int grp = k4 >> 3;
    int h0 = 2 * (((k4 & 7) >> 1) + (hi ? 4 : 0));
    size_t base = ((size_t)(tm * (K >> 5) + tk)) * 512 + (size_t)(l16 + 16 * grp) * 16 + h0;
    h4 o;
    o.x = (_Float16)v.x; o.y = (_Float16)v.y; o.z = (_Float16)v.z; o.w = (_Float16)v.w;
    *reinterpret_cast<h4*>(&P[base]) = o;
}

// B operand: per (n-tile of 16, k-tile of 32) block of 512 halfs:
//   offset = (tn*(K/32)+tk)*512 + (n%16 + 16*((k&15)>>3))*16 + (k&7) + 8*((k&31)>>4)
__global__ void k_pack_b(const float* __restrict__ W, _Float16* __restrict__ P,
                         int K, int Nc) {
    int i = blockIdx.x * blockDim.x + threadIdx.x;
    int total = (K >> 2) * Nc;
    if (i >= total) return;
    int kgrp = i / Nc;
    int n = i - kgrp * Nc;
    int k0 = kgrp << 2;
    float a = W[(size_t)(k0 + 0) * Nc + n];
    float b = W[(size_t)(k0 + 1) * Nc + n];
    float c = W[(size_t)(k0 + 2) * Nc + n];
    float d = W[(size_t)(k0 + 3) * Nc + n];
    int tn = n >> 4, l16 = n & 15;
    int tk = k0 >> 5, kk = k0 & 31, grp = (kk & 15) >> 3, h0 = (kk & 7) + ((kk >> 4) << 3);
    size_t base = ((size_t)(tn * (K >> 5) + tk)) * 512 + (size_t)(l16 + 16 * grp) * 16 + h0;
    h4 o;
    o.x = (_Float16)a; o.y = (_Float16)b; o.z = (_Float16)c; o.w = (_Float16)d;
    *reinterpret_cast<h4*>(&P[base]) = o;
}

// ---------------------------------------------------------------------------
// Fused dual-product WMMA GEMM on packed operands:
//   C = relu(A0@B0 + A1@B1 + bias)
// 128x64 tile per block, 128 threads = 4 waves. Each wave owns 32 rows
// (2 A fragments, read straight from global: contiguous 32B/lane,
// L2-resident across N-blocks) x 64 cols (4 B fragments staged in LDS,
// async double-buffered) -> 8 WMMAs per K-step per wave.
// ---------------------------------------------------------------------------
__global__ __launch_bounds__(128)
void k_gemm_packed(const _Float16* __restrict__ A0p, const _Float16* __restrict__ B0p, int K0,
                   const _Float16* __restrict__ A1p, const _Float16* __restrict__ B1p, int K1,
                   const float* __restrict__ bias, float* __restrict__ C,
                   int M, int Nc) {
    __shared__ _Float16 Bs[2][4 * 512];     // 2 x 4KB double buffer

    const int tid  = threadIdx.x;
    const int lane = tid & 31;
    const int wv   = tid >> 5;
    const int bm   = blockIdx.y * 128;
    const int bn   = blockIdx.x * 64;
    const int grp  = lane >> 4;
    const int l16  = lane & 15;

    const int steps0 = K0 >> 5;
    const int steps1 = K1 >> 5;
    const int total  = steps0 + steps1;

    v8f acc[2][4];
    #pragma unroll
    for (int mt = 0; mt < 2; ++mt)
        #pragma unroll
        for (int t = 0; t < 4; ++t)
            #pragma unroll
            for (int j = 0; j < 8; ++j) acc[mt][t][j] = 0.0f;

    auto params = [&](int s, const _Float16*& Ap, const _Float16*& Bp,
                      int& nKt, int& tk) {
        if (s < steps0) { Ap = A0p; Bp = B0p; nKt = steps0; tk = s; }
        else            { Ap = A1p; Bp = B1p; nKt = steps1; tk = s - steps0; }
    };

    // stage 4 B-tiles (4KB) for step (Bp,nKt,tk) into buffer `buf`
    auto stage = [&](const _Float16* Bp, int nKt, int tk, int buf) {
        for (int c = tid; c < 256; c += 128) {           // 256 x 16B chunks
            int nt = c >> 6, ch = c & 63;
            int tn = blockIdx.x * 4 + nt;
            const _Float16* src = Bp + ((size_t)(tn * nKt + tk)) * 512 + ch * 8;
            _Float16* dst = &Bs[buf][nt * 512 + ch * 8];
#if HAVE_ASYNC_LDS
            __builtin_amdgcn_global_load_async_to_lds_b128(
                (__attribute__((address_space(1))) i4*)src,
                (__attribute__((address_space(3))) i4*)dst, 0, 0);
#else
            *reinterpret_cast<uint4*>(dst) = *reinterpret_cast<const uint4*>(src);
#endif
        }
    };

    auto compute = [&](const _Float16* Ap, int nKt, int tk, int buf) {
        const int tm0 = blockIdx.y * 8 + wv * 2;         // 2 consecutive m-tiles
        const _Float16* abase = Ap + ((size_t)(tm0 * nKt + tk)) * 512 + lane * 16;
        v16h afrag0 = *reinterpret_cast<const v16h*>(abase);
        v16h afrag1 = *reinterpret_cast<const v16h*>(abase + (size_t)nKt * 512);
        if (tk + 1 < nKt) {                              // global_prefetch_b8
            __builtin_prefetch(abase + 512, 0, 1);
            __builtin_prefetch(abase + (size_t)nKt * 512 + 512, 0, 1);
        }
        #pragma unroll
        for (int nt = 0; nt < 4; ++nt) {
            v16h bfrag = *reinterpret_cast<const v16h*>(&Bs[buf][nt * 512 + lane * 16]);
            acc[0][nt] = __builtin_amdgcn_wmma_f32_16x16x32_f16(
                false, afrag0, false, bfrag, (short)0, acc[0][nt], false, false);
            acc[1][nt] = __builtin_amdgcn_wmma_f32_16x16x32_f16(
                false, afrag1, false, bfrag, (short)0, acc[1][nt], false, false);
        }
    };

#if HAVE_ASYNC_LDS
    {
        const _Float16 *Ap, *Bp; int nKt, tk;
        params(0, Ap, Bp, nKt, tk);
        stage(Bp, nKt, tk, 0);
    }
    for (int s = 0; s < total; ++s) {
        int buf = s & 1;
#if defined(__has_builtin) && __has_builtin(__builtin_amdgcn_s_wait_asynccnt)
        __builtin_amdgcn_s_wait_asynccnt(0);
#else
        asm volatile("s_wait_asynccnt 0x0" ::: "memory");
#endif
        __syncthreads();
        if (s + 1 < total) {
            const _Float16 *Ap, *Bp; int nKt, tk;
            params(s + 1, Ap, Bp, nKt, tk);
            stage(Bp, nKt, tk, buf ^ 1);
        }
        const _Float16 *Ap, *Bp; int nKt, tk;
        params(s, Ap, Bp, nKt, tk);
        compute(Ap, nKt, tk, buf);
        __syncthreads();
    }
#else
    for (int s = 0; s < total; ++s) {
        const _Float16 *Ap, *Bp; int nKt, tk;
        params(s, Ap, Bp, nKt, tk);
        stage(Bp, nKt, tk, 0);
        __syncthreads();
        compute(Ap, nKt, tk, 0);
        __syncthreads();
    }
#endif

    // Epilogue: bias + relu. C/D layout: float r of lane -> M offset r + 8*grp, N = l16
    #pragma unroll
    for (int mt = 0; mt < 2; ++mt) {
        #pragma unroll
        for (int nt = 0; nt < 4; ++nt) {
            int ncol = bn + nt * 16 + l16;
            float bv = bias ? bias[ncol] : 0.0f;
            #pragma unroll
            for (int r = 0; r < 8; ++r) {
                int mrow = bm + (wv * 2 + mt) * 16 + r + 8 * grp;
                float v = acc[mt][nt][r] + bv;
                C[(size_t)mrow * Nc + ncol] = v > 0.0f ? v : 0.0f;
            }
        }
    }
}

// ---------------------------------------------------------------------------
// Head kernels
// ---------------------------------------------------------------------------
__global__ __launch_bounds__(256)
void k_matvec_rowsum(const float* __restrict__ x2, const float* __restrict__ Wa,
                     const float* __restrict__ ba, float* __restrict__ scores,
                     float* __restrict__ rowsum, int F) {
    __shared__ float rp[256];
    __shared__ float rr[256];
    int n = blockIdx.x, tid = threadIdx.x;
    float p = 0.0f, s = 0.0f;
    for (int f = tid; f < F; f += 256) {
        float v = x2[(size_t)n * F + f];
        p += v * Wa[f];
        s += v;
    }
    rp[tid] = p; rr[tid] = s;
    __syncthreads();
    for (int o = 128; o > 0; o >>= 1) {
        if (tid < o) { rp[tid] += rp[tid + o]; rr[tid] += rr[tid + o]; }
        __syncthreads();
    }
    if (tid == 0) {
        scores[n] = rp[0] + ba[0];
        rowsum[n] = rr[0];
    }
}

__global__ void k_colmean(const float* __restrict__ x2, float* __restrict__ pooled,
                          int Nn, int F) {
    int c = blockIdx.x * blockDim.x + threadIdx.x;
    if (c >= F) return;
    float s = 0.0f;
    for (int n = 0; n < Nn; ++n) s += x2[(size_t)n * F + c];
    pooled[c] = s / (float)Nn;
}

__global__ __launch_bounds__(1024)
void k_softmax_ge(const float* __restrict__ scores, const float* __restrict__ rowsum,
                  float* __restrict__ ge, int Nn) {
    __shared__ float red[1024];
    int tid = threadIdx.x;
    float m = -3.402823466e38f;
    for (int i = tid; i < Nn; i += 1024) m = fmaxf(m, scores[i]);
    red[tid] = m; __syncthreads();
    for (int o = 512; o > 0; o >>= 1) {
        if (tid < o) red[tid] = fmaxf(red[tid], red[tid + o]);
        __syncthreads();
    }
    float mx = red[0];
    __syncthreads();
    float s = 0.0f;
    for (int i = tid; i < Nn; i += 1024) s += expf(scores[i] - mx);
    red[tid] = s; __syncthreads();
    for (int o = 512; o > 0; o >>= 1) {
        if (tid < o) red[tid] += red[tid + o];
        __syncthreads();
    }
    float tot = red[0];
    for (int i = tid; i < Nn; i += 1024)
        ge[i] = (expf(scores[i] - mx) / tot) * rowsum[i];
}

__global__ __launch_bounds__(256)
void k_final(const float* __restrict__ pooled, const float* __restrict__ ge,
             const float* __restrict__ Wf, const float* __restrict__ bf,
             float* __restrict__ out, int F2) {
    __shared__ float red[256];
    int tid = threadIdx.x;
    float acc[NCLS];
    #pragma unroll
    for (int c = 0; c < NCLS; ++c) acc[c] = 0.0f;
    int half = F2 >> 1;
    for (int k = tid; k < F2; k += 256) {
        float v = (k < half) ? pooled[k] : ge[k - half];
        #pragma unroll
        for (int c = 0; c < NCLS; ++c) acc[c] += v * Wf[(size_t)k * NCLS + c];
    }
    for (int c = 0; c < NCLS; ++c) {
        red[tid] = acc[c];
        __syncthreads();
        for (int o = 128; o > 0; o >>= 1) {
            if (tid < o) red[tid] += red[tid + o];
            __syncthreads();
        }
        if (tid == 0) out[c] = red[0] + bf[c];
        __syncthreads();
    }
}

// ---------------------------------------------------------------------------
// Launch
// ---------------------------------------------------------------------------
extern "C" void kernel_launch(void* const* d_in, const int* in_sizes, int n_in,
                              void* d_out, int out_size, void* d_ws, size_t ws_size,
                              hipStream_t stream) {
    const float* x   = (const float*)d_in[0];
    const int*   src = (const int*)  d_in[1];
    const int*   dst = (const int*)  d_in[2];
    const float* W1l = (const float*)d_in[3];
    const float* b1  = (const float*)d_in[4];
    const float* W1r = (const float*)d_in[5];
    const float* W2l = (const float*)d_in[6];
    const float* b2  = (const float*)d_in[7];
    const float* W2r = (const float*)d_in[8];
    const float* Wa  = (const float*)d_in[9];
    const float* ba  = (const float*)d_in[10];
    const float* Wf  = (const float*)d_in[11];
    const float* bf  = (const float*)d_in[12];
    float* out = (float*)d_out;

    // workspace carve-up
    float* ws     = (float*)d_ws;
    float* deg    = ws;                               // NODES
    float* sum1   = deg    + NODES;                   // NODES*INDIM
    float* x1     = sum1   + (size_t)NODES * INDIM;   // NODES*HIDD
    float* sum2   = x1     + (size_t)NODES * HIDD;    // NODES*HIDD
    float* x2     = sum2   + (size_t)NODES * HIDD;    // NODES*HIDD
    float* scores = x2     + (size_t)NODES * HIDD;    // NODES
    float* rowsum = scores + NODES;                   // NODES
    float* pooled = rowsum + NODES;                   // HIDD
    float* ge     = pooled + HIDD;                    // NODES
    _Float16* A0p = (_Float16*)(ge + NODES);          // NODES*HIDD halfs (max)
    _Float16* A1p = A0p + (size_t)NODES * HIDD;
    _Float16* B0p = A1p + (size_t)NODES * HIDD;       // HIDD*HIDD halfs (max)
    _Float16* B1p = B0p + (size_t)HIDD * HIDD;

    (void)hipMemsetAsync(deg,  0, NODES * sizeof(float), stream);
    (void)hipMemsetAsync(sum1, 0, (size_t)NODES * INDIM * sizeof(float), stream);
    (void)hipMemsetAsync(sum2, 0, (size_t)NODES * HIDD * sizeof(float), stream);

    k_degree<<<(EDGES + 255) / 256, 256, 0, stream>>>(dst, deg, EDGES);

    // ----- layer 1 -----
    {
        int total = EDGES * (INDIM / 4);
        k_scatter_add<<<(total + 255) / 256, 256, 0, stream>>>(x, src, dst, sum1, EDGES, INDIM);
        int nf = NODES * INDIM;
        k_scale<<<(nf + 255) / 256, 256, 0, stream>>>(sum1, deg, NODES, INDIM);

        int pa = NODES * INDIM / 4;
        k_pack_a<<<(pa + 255) / 256, 256, 0, stream>>>(sum1, A0p, NODES, INDIM);
        k_pack_a<<<(pa + 255) / 256, 256, 0, stream>>>(x,    A1p, NODES, INDIM);
        int pb = (INDIM / 4) * HIDD;
        k_pack_b<<<(pb + 255) / 256, 256, 0, stream>>>(W1l, B0p, INDIM, HIDD);
        k_pack_b<<<(pb + 255) / 256, 256, 0, stream>>>(W1r, B1p, INDIM, HIDD);

        dim3 grid(HIDD / 64, NODES / 128);
        k_gemm_packed<<<grid, 128, 0, stream>>>(A0p, B0p, INDIM,
                                                A1p, B1p, INDIM,
                                                b1, x1, NODES, HIDD);
    }

    // ----- layer 2 -----
    {
        int total = EDGES * (HIDD / 4);
        k_scatter_add<<<(total + 255) / 256, 256, 0, stream>>>(x1, src, dst, sum2, EDGES, HIDD);
        int nf = NODES * HIDD;
        k_scale<<<(nf + 255) / 256, 256, 0, stream>>>(sum2, deg, NODES, HIDD);

        int pa = NODES * HIDD / 4;
        k_pack_a<<<(pa + 255) / 256, 256, 0, stream>>>(sum2, A0p, NODES, HIDD);
        k_pack_a<<<(pa + 255) / 256, 256, 0, stream>>>(x1,   A1p, NODES, HIDD);
        int pb = (HIDD / 4) * HIDD;
        k_pack_b<<<(pb + 255) / 256, 256, 0, stream>>>(W2l, B0p, HIDD, HIDD);
        k_pack_b<<<(pb + 255) / 256, 256, 0, stream>>>(W2r, B1p, HIDD, HIDD);

        dim3 grid(HIDD / 64, NODES / 128);
        k_gemm_packed<<<grid, 128, 0, stream>>>(A0p, B0p, HIDD,
                                                A1p, B1p, HIDD,
                                                b2, x2, NODES, HIDD);
    }

    // ----- head -----
    k_matvec_rowsum<<<NODES, 256, 0, stream>>>(x2, Wa, ba, scores, rowsum, HIDD);
    k_colmean<<<(HIDD + 255) / 256, 256, 0, stream>>>(x2, pooled, NODES, HIDD);
    k_softmax_ge<<<1, 1024, 0, stream>>>(scores, rowsum, ge, NODES);
    k_final<<<1, 256, 0, stream>>>(pooled, ge, Wf, bf, out, 2 * HIDD);
}
